// AttentionBlock_59476707115137
// MI455X (gfx1250) — matrix operand order repaired and checked by reading, not compiled, not run on hardware
//
#include <hip/hip_runtime.h>
#include <hip/hip_bf16.h>
#include <stddef.h>

// Problem constants (from reference)
#define BB   2
#define TT   2048
#define EE   768
#define HH   12
#define DD   64
#define HID  3072
#define QKVN 2304           // 3*E combined qkv columns
#define MROWS (BB*TT)       // 4096

typedef __attribute__((ext_vector_type(16))) __bf16 v16bf;
typedef __attribute__((ext_vector_type(8)))  float  v8f;
typedef __attribute__((ext_vector_type(4)))  int    v4i;
typedef __attribute__((address_space(1))) v4i g_v4i;   // global int4
typedef __attribute__((address_space(3))) v4i l_v4i;   // LDS int4

union Frag16 { uint4 u[2]; v16bf v; };

__device__ __forceinline__ unsigned short f2bf(float f) {
    unsigned int u = __float_as_uint(f);
    unsigned int r = u + 0x7FFFu + ((u >> 16) & 1u);   // round to nearest even
    return (unsigned short)(r >> 16);
}

__device__ __forceinline__ v8f wmma_bf16(v16bf a, v16bf b, v8f c) {
    return __builtin_amdgcn_wmma_f32_16x16x32_bf16(
        /*neg_a=*/false, a, /*neg_b=*/false, b,
        /*c_mod=*/(short)0, c, /*reuse_a=*/false, /*reuse_b=*/false);
}

// ---- CDNA5 async copy: global -> LDS, tracked by ASYNCcnt --------------------
__device__ __forceinline__ void async_copy_b128(const unsigned short* gsrc,
                                                unsigned short* lds_dst) {
#if __has_builtin(__builtin_amdgcn_global_load_async_to_lds_b128)
    __builtin_amdgcn_global_load_async_to_lds_b128(
        (g_v4i*)gsrc, (l_v4i*)lds_dst, /*imm offset=*/0, /*cpol=*/0);
#else
    *(uint4*)lds_dst = *(const uint4*)gsrc;
#endif
}

__device__ __forceinline__ void wait_async0() {
#if __has_builtin(__builtin_amdgcn_s_wait_asynccnt)
    __builtin_amdgcn_s_wait_asynccnt(0);
#else
    asm volatile("s_wait_asynccnt 0" ::: "memory");
#endif
}

// ---- fragment loads from LDS tiles ------------------------------------------
// A-fragment (16x32 bf16, MxK), tile row stride = 32 bf16.
// lanes 0-15: row M=lane, K {0..7,16..23}; lanes 16-31: row M=lane-16, K {8..15,24..31}.
__device__ __forceinline__ v16bf lds_a_frag(const unsigned short* base) {
    const int lane = threadIdx.x & 31;
    const unsigned short* p = base + (lane & 15) * 32 + ((lane < 16) ? 0 : 8);
    Frag16 f;
    f.u[0] = *(const uint4*)(p);
    f.u[1] = *(const uint4*)(p + 16);
    return f.v;
}

// B-fragment (32x16 bf16, KxN) from a [N][K]-major LDS tile with row stride ldk.
// lane = column N (lane&15); lanes 0-15 hold K {0..7,16..23}, lanes 16-31 K {8..15,24..31}.
__device__ __forceinline__ v16bf lds_b_frag(const unsigned short* base, int col0,
                                            int ldk, int k0) {
    const int lane = threadIdx.x & 31;
    const unsigned short* p =
        base + (col0 + (lane & 15)) * ldk + k0 + ((lane < 16) ? 0 : 8);
    Frag16 f;
    f.u[0] = *(const uint4*)(p);
    f.u[1] = *(const uint4*)(p + 16);
    return f.v;
}

// Direct-from-global A-fragment load
__device__ __forceinline__ v16bf load_a_frag(const unsigned short* A, int lda,
                                             int row0, int k0) {
    const int lane = threadIdx.x & 31;
    const unsigned short* p =
        A + (size_t)(row0 + (lane & 15)) * lda + k0 + ((lane < 16) ? 0 : 8);
    Frag16 f;
    f.u[0] = *(const uint4*)(p);
    f.u[1] = *(const uint4*)(p + 16);
    return f.v;
}

// ---------------------------------------------------------------- pack kernels
__global__ void k_cvt_bf16(const float* __restrict__ in,
                           unsigned short* __restrict__ out, int n) {
    int i = blockIdx.x * blockDim.x + threadIdx.x;
    if (i < n) out[i] = f2bf(in[i]);
}

// Wq/Wk/Wv [H,E,D] fp32 -> Wqkv_t [2304][768] bf16 (row j = output column)
__global__ void k_pack_qkv(const float* __restrict__ Wq,
                           const float* __restrict__ Wk,
                           const float* __restrict__ Wv,
                           unsigned short* __restrict__ out) {
    int i = blockIdx.x * blockDim.x + threadIdx.x;  // over 2304*768
    if (i >= QKVN * EE) return;
    int j = i / EE;        // combined column 0..2303
    int e = i % EE;
    int which = j / EE;    // 0=q,1=k,2=v
    int jj = j % EE;
    int h = jj / DD, d = jj % DD;
    const float* W = (which == 0) ? Wq : (which == 1) ? Wk : Wv;
    out[(size_t)j * EE + e] = f2bf(W[((size_t)h * EE + e) * DD + d]);
}

// W1 [768][3072] -> W1t [3072][768]
__global__ void k_pack_w1(const float* __restrict__ W1,
                          unsigned short* __restrict__ out) {
    int i = blockIdx.x * blockDim.x + threadIdx.x;  // over 3072*768
    if (i >= HID * EE) return;
    int n = i / EE, e = i % EE;
    out[(size_t)n * EE + e] = f2bf(W1[(size_t)e * HID + n]);
}

// W2 [3072][768] -> W2t [768][3072]
__global__ void k_pack_w2(const float* __restrict__ W2,
                          unsigned short* __restrict__ out) {
    int i = blockIdx.x * blockDim.x + threadIdx.x;  // over 768*3072
    if (i >= EE * HID) return;
    int e = i / HID, n = i % HID;
    out[(size_t)e * HID + n] = f2bf(W2[(size_t)n * EE + e]);
}

// qkv bf16 [B*T,2304] -> Vt bf16 [B,H,D,T]
__global__ void k_transpose_v(const unsigned short* __restrict__ qkv,
                              unsigned short* __restrict__ vt) {
    int i = blockIdx.x * blockDim.x + threadIdx.x;  // over B*H*D*T
    if (i >= BB * HH * DD * TT) return;
    int t = i % TT;
    int rest = i / TT;
    int d = rest % DD;
    int bh = rest / DD;
    int b = bh / HH, h = bh % HH;
    vt[i] = qkv[((size_t)b * TT + t) * QKVN + 2 * EE + h * DD + d];
}

// ---------------------------------------------------------------- WMMA GEMM
// C[M,N] = A[M,K] * Bt[N,K]^T (+bias, relu). 256 thr = 8 waves.
// Block tile 64x128; K staged 32-wide into double-buffered LDS via async DMA.
template <int OUT_BF16>
__global__ void k_gemm(const unsigned short* __restrict__ A,
                       const unsigned short* __restrict__ Bt,
                       const float* __restrict__ bias,
                       float* __restrict__ Cf,
                       unsigned short* __restrict__ Cb,
                       int M, int N, int K, int relu) {
    __shared__ __align__(16) unsigned short ldsA[2][64 * 32];   //  4KB x2
    __shared__ __align__(16) unsigned short ldsB[2][128 * 32];  //  8KB x2
    const int tid  = threadIdx.x;
    const int wave = tid >> 5;
    const int lane = tid & 31;
    const int lo   = lane & 15;
    const int hi8  = (lane < 16) ? 0 : 8;
    const int brow = blockIdx.x * 64;
    const int bcol = blockIdx.y * 128;
    const int wrowL = (wave & 3) * 16;   // row offset inside block tile
    const int wcolL = (wave >> 2) * 64;  // col offset inside block tile

    // async-DMA one 32-wide K stage into LDS buffer `buf`
    auto copy_stage = [&](int k, int buf) {
        // A tile: 64 rows x 32 bf16 = 256 x 16B chunks, 1 per thread
        {
            int row = tid >> 2, q = tid & 3;
            async_copy_b128(A + (size_t)(brow + row) * K + k + q * 8,
                            &ldsA[buf][row * 32 + q * 8]);
        }
        // B tile: 128 rows x 32 bf16 = 512 x 16B chunks, 2 per thread
        #pragma unroll
        for (int c = 0; c < 2; ++c) {
            int ch = tid + c * 256;
            int n = ch >> 2, q = ch & 3;
            async_copy_b128(Bt + (size_t)(bcol + n) * K + k + q * 8,
                            &ldsB[buf][n * 32 + q * 8]);
        }
    };

    v8f acc0 = {}, acc1 = {}, acc2 = {}, acc3 = {};
    copy_stage(0, 0);
    wait_async0();
    __syncthreads();
    int buf = 0;
    for (int k = 0; k < K; k += 32) {
        if (k + 32 < K) copy_stage(k + 32, buf ^ 1);   // overlaps with WMMAs below
        // hoist all fragment loads so LDS reads pipeline ahead of the WMMA chain
        v16bf a  = lds_a_frag(&ldsA[buf][wrowL * 32]);
        v16bf b0 = lds_b_frag(ldsB[buf], wcolL +  0, 32, 0);
        v16bf b1 = lds_b_frag(ldsB[buf], wcolL + 16, 32, 0);
        v16bf b2 = lds_b_frag(ldsB[buf], wcolL + 32, 32, 0);
        v16bf b3 = lds_b_frag(ldsB[buf], wcolL + 48, 32, 0);
        acc0 = wmma_bf16(a, b0, acc0);
        acc1 = wmma_bf16(a, b1, acc1);
        acc2 = wmma_bf16(a, b2, acc2);
        acc3 = wmma_bf16(a, b3, acc3);
        wait_async0();
        __syncthreads();
        buf ^= 1;
    }
    #pragma unroll
    for (int j = 0; j < 4; ++j) {
        v8f acc = (j == 0) ? acc0 : (j == 1) ? acc1 : (j == 2) ? acc2 : acc3;
        int col = bcol + wcolL + 16 * j + lo;
        float bv = bias ? bias[col] : 0.0f;
        #pragma unroll
        for (int r = 0; r < 8; ++r) {
            int row = brow + wrowL + hi8 + r;
            float v = acc[r] + bv;
            if (relu) v = fmaxf(v, 0.0f);
            if (OUT_BF16) Cb[(size_t)row * N + col] = f2bf(v);
            else          Cf[(size_t)row * N + col] = v;
        }
    }
}

// ---------------------------------------------------------------- attention
// Grid: B*H*(T/64) blocks, 128 threads (4 waves); wave owns 16 q rows.
// K/V chunks (32 keys) staged via async DMA, double buffered, shared by waves.
__global__ void k_attention(const unsigned short* __restrict__ qkv,
                            const unsigned short* __restrict__ vt,
                            float* __restrict__ attnout) {
    __shared__ __align__(16) unsigned short ldsK[2][32 * 64];  // [s][d] 4KB x2
    __shared__ __align__(16) unsigned short ldsV[2][64 * 32];  // [d][s] 4KB x2
    __shared__ __align__(16) unsigned short pTile[4][16 * 32];
    const int tid  = threadIdx.x;
    const int wave = tid >> 5;
    const int lane = tid & 31;
    const int lo   = lane & 15;
    const int hi8  = (lane < 16) ? 0 : 8;

    const int blocksPerBH = TT / 64;
    const int bh = blockIdx.x / blocksPerBH;
    const int qb = blockIdx.x % blocksPerBH;
    const int b = bh / HH, h = bh % HH;
    const int t0 = qb * 64 + wave * 16;   // this wave's first q row

    const unsigned short* Aq = qkv + (size_t)b * TT * QKVN + h * DD;
    const unsigned short* Bk = qkv + (size_t)b * TT * QKVN + EE + h * DD;
    const unsigned short* Bv = vt + (size_t)bh * DD * TT;

    const v16bf aq0 = load_a_frag(Aq, QKVN, t0, 0);
    const v16bf aq1 = load_a_frag(Aq, QKVN, t0, 32);

    // stage K rows (64 x 32 bf16) + Vt rows (64 x 32 bf16), 2 chunks/thread each
    auto copy_stage = [&](int s0, int buf) {
        #pragma unroll
        for (int c = 0; c < 2; ++c) {
            int ch = tid + c * 128;
            int s = ch >> 2, q = ch & 3;    // 64 rows x 4 chunks
            async_copy_b128(Bk + (size_t)(s0 + s) * QKVN + q * 8,
                            &ldsK[buf][s * 64 + q * 8]);
        }
        #pragma unroll
        for (int c = 0; c < 2; ++c) {
            int ch = tid + c * 128;
            int d = ch >> 2, q = ch & 3;    // 64 rows x 4 chunks
            async_copy_b128(Bv + (size_t)d * TT + s0 + q * 8,
                            &ldsV[buf][d * 32 + q * 8]);
        }
    };

    v8f o0 = {}, o1 = {}, o2 = {}, o3 = {};
    float mrow[8], lrow[8];
    #pragma unroll
    for (int r = 0; r < 8; ++r) { mrow[r] = -1e30f; lrow[r] = 0.0f; }

    const int tmax    = t0 + 15;       // wave causal bound
    const int smaxblk = qb * 64 + 63;  // block-uniform loop bound
    copy_stage(0, 0);
    wait_async0();
    __syncthreads();
    int buf = 0;
    for (int s0 = 0; s0 <= smaxblk; s0 += 32) {
        if (s0 + 32 <= smaxblk) copy_stage(s0 + 32, buf ^ 1);
        if (s0 <= tmax) {  // wave-uniform predicate
            // hoist all K fragments, then issue the QK^T WMMA chain
            v16bf bk00 = lds_b_frag(ldsK[buf],  0, 64,  0);
            v16bf bk01 = lds_b_frag(ldsK[buf],  0, 64, 32);
            v16bf bk10 = lds_b_frag(ldsK[buf], 16, 64,  0);
            v16bf bk11 = lds_b_frag(ldsK[buf], 16, 64, 32);
            v8f c0 = {}, c1 = {};
            c0 = wmma_bf16(aq0, bk00, c0);
            c0 = wmma_bf16(aq1, bk01, c0);
            c1 = wmma_bf16(aq0, bk10, c1);
            c1 = wmma_bf16(aq1, bk11, c1);

            float pv[2][8];
            #pragma unroll
            for (int tile = 0; tile < 2; ++tile) {
                const v8f c = tile ? c1 : c0;
                const int scol = s0 + 16 * tile + lo;
                #pragma unroll
                for (int r = 0; r < 8; ++r) {
                    int trow = t0 + hi8 + r;
                    pv[tile][r] = (scol <= trow) ? c[r] * 0.125f : -1e30f;
                }
            }
            float newm[8], scale[8];
            #pragma unroll
            for (int r = 0; r < 8; ++r) {
                float tm = fmaxf(pv[0][r], pv[1][r]);
                tm = fmaxf(tm, __shfl_xor(tm, 1, 32));
                tm = fmaxf(tm, __shfl_xor(tm, 2, 32));
                tm = fmaxf(tm, __shfl_xor(tm, 4, 32));
                tm = fmaxf(tm, __shfl_xor(tm, 8, 32));
                newm[r] = fmaxf(mrow[r], tm);
                scale[r] = __expf(mrow[r] - newm[r]);
            }
            #pragma unroll
            for (int tile = 0; tile < 2; ++tile) {
                #pragma unroll
                for (int r = 0; r < 8; ++r) {
                    float p = __expf(pv[tile][r] - newm[r]);
                    pv[tile][r] = p;
                    pTile[wave][(hi8 + r) * 32 + 16 * tile + lo] = f2bf(p);
                }
            }
            #pragma unroll
            for (int r = 0; r < 8; ++r) {
                float rs = pv[0][r] + pv[1][r];
                rs += __shfl_xor(rs, 1, 32);
                rs += __shfl_xor(rs, 2, 32);
                rs += __shfl_xor(rs, 4, 32);
                rs += __shfl_xor(rs, 8, 32);
                lrow[r] = lrow[r] * scale[r] + rs;
                mrow[r] = newm[r];
                o0[r] *= scale[r]; o1[r] *= scale[r];
                o2[r] *= scale[r]; o3[r] *= scale[r];
            }
            // per-wave DS in-order + explicit split-counter wait, then relayout P
            asm volatile("s_wait_dscnt 0" ::: "memory");
            const unsigned short* pp = &pTile[wave][lo * 32 + ((lane < 16) ? 0 : 8)];
            Frag16 pf;
            pf.u[0] = *(const uint4*)(pp);
            pf.u[1] = *(const uint4*)(pp + 16);
            const v16bf pa = pf.v;
            v16bf bv0 = lds_b_frag(ldsV[buf],  0, 32, 0);
            v16bf bv1 = lds_b_frag(ldsV[buf], 16, 32, 0);
            v16bf bv2 = lds_b_frag(ldsV[buf], 32, 32, 0);
            v16bf bv3 = lds_b_frag(ldsV[buf], 48, 32, 0);
            o0 = wmma_bf16(pa, bv0, o0);
            o1 = wmma_bf16(pa, bv1, o1);
            o2 = wmma_bf16(pa, bv2, o2);
            o3 = wmma_bf16(pa, bv3, o3);
        }
        wait_async0();
        __syncthreads();
        buf ^= 1;
    }
    #pragma unroll
    for (int r = 0; r < 8; ++r) {
        float inv = 1.0f / lrow[r];
        size_t rowoff = (size_t)(b * TT + t0 + hi8 + r) * EE + h * DD;
        attnout[rowoff +  0 + lo] = o0[r] * inv;
        attnout[rowoff + 16 + lo] = o1[r] * inv;
        attnout[rowoff + 32 + lo] = o2[r] * inv;
        attnout[rowoff + 48 + lo] = o3[r] * inv;
    }
}

// ------------------------------------------------------- residual + layernorm
__global__ void k_ln(const float* __restrict__ a, const float* __restrict__ c,
                     const float* __restrict__ g, const float* __restrict__ beta,
                     float* __restrict__ yout, unsigned short* __restrict__ ybout) {
    __shared__ float s1[256], s2[256];
    const int row = blockIdx.x;
    const int tid = threadIdx.x;
    const float* pa = a + (size_t)row * EE;
    const float* pc = c + (size_t)row * EE;
    float v[3], sum = 0.0f, sq = 0.0f;
    #pragma unroll
    for (int i = 0; i < 3; ++i) {
        int col = tid + i * 256;
        v[i] = pa[col] + pc[col];
        sum += v[i];
        sq += v[i] * v[i];
    }
    s1[tid] = sum; s2[tid] = sq;
    __syncthreads();
    for (int off = 128; off > 0; off >>= 1) {
        if (tid < off) { s1[tid] += s1[tid + off]; s2[tid] += s2[tid + off]; }
        __syncthreads();
    }
    float mu = s1[0] * (1.0f / EE);
    float var = s2[0] * (1.0f / EE) - mu * mu;
    float rs = rsqrtf(var + 1e-5f);
    #pragma unroll
    for (int i = 0; i < 3; ++i) {
        int col = tid + i * 256;
        float y = (v[i] - mu) * rs * g[col] + beta[col];
        yout[(size_t)row * EE + col] = y;
        if (ybout) ybout[(size_t)row * EE + col] = f2bf(y);
    }
}

// -------------------------------------------------------------------- launch
extern "C" void kernel_launch(void* const* d_in, const int* in_sizes, int n_in,
                              void* d_out, int out_size, void* d_ws, size_t ws_size,
                              hipStream_t stream) {
    const float* x     = (const float*)d_in[0];
    const float* Wq    = (const float*)d_in[1];
    const float* Wk    = (const float*)d_in[2];
    const float* Wv    = (const float*)d_in[3];
    const float* W1    = (const float*)d_in[4];
    const float* b1    = (const float*)d_in[5];
    const float* W2    = (const float*)d_in[6];
    const float* b2    = (const float*)d_in[7];
    const float* ln1_g = (const float*)d_in[8];
    const float* ln1_b = (const float*)d_in[9];
    const float* ln2_g = (const float*)d_in[10];
    const float* ln2_b = (const float*)d_in[11];
    float* out = (float*)d_out;

    char* ws = (char*)d_ws;
    size_t off = 0;
    unsigned short* xb    = (unsigned short*)(ws + off); off += (size_t)MROWS * EE * 2;
    unsigned short* wqkvt = (unsigned short*)(ws + off); off += (size_t)QKVN * EE * 2;
    unsigned short* w1t   = (unsigned short*)(ws + off); off += (size_t)HID * EE * 2;
    unsigned short* w2t   = (unsigned short*)(ws + off); off += (size_t)EE * HID * 2;
    unsigned short* qkvb  = (unsigned short*)(ws + off); off += (size_t)MROWS * QKVN * 2;
    unsigned short* vtb   = (unsigned short*)(ws + off); off += (size_t)BB * HH * DD * TT * 2;
    float*          attnf = (float*)(ws + off);          off += (size_t)MROWS * EE * 4;
    float*          yf    = (float*)(ws + off);          off += (size_t)MROWS * EE * 4;
    unsigned short* yb    = (unsigned short*)(ws + off); off += (size_t)MROWS * EE * 2;
    unsigned short* hb    = (unsigned short*)(ws + off); off += (size_t)MROWS * HID * 2;
    float*          mlpf  = (float*)(ws + off);          off += (size_t)MROWS * EE * 4;
    (void)ws_size; (void)in_sizes; (void)n_in; (void)out_size;

    {
        int n = MROWS * EE;
        k_cvt_bf16<<<(n + 255) / 256, 256, 0, stream>>>(x, xb, n);
    }
    k_pack_qkv<<<(QKVN * EE + 255) / 256, 256, 0, stream>>>(Wq, Wk, Wv, wqkvt);
    k_pack_w1<<<(HID * EE + 255) / 256, 256, 0, stream>>>(W1, w1t);
    k_pack_w2<<<(EE * HID + 255) / 256, 256, 0, stream>>>(W2, w2t);

    // fused QKV projection: [4096,768] x [768,2304] -> bf16
    k_gemm<1><<<dim3(MROWS / 64, QKVN / 128), 256, 0, stream>>>(
        xb, wqkvt, nullptr, nullptr, qkvb, MROWS, QKVN, EE, 0);

    k_transpose_v<<<(BB * HH * DD * TT + 255) / 256, 256, 0, stream>>>(qkvb, vtb);

    k_attention<<<BB * HH * (TT / 64), 128, 0, stream>>>(qkvb, vtb, attnf);

    k_ln<<<MROWS, 256, 0, stream>>>(x, attnf, ln1_g, ln1_b, yf, yb);

    // MLP up: relu(y @ W1 + b1) -> bf16 [4096,3072]
    k_gemm<1><<<dim3(MROWS / 64, HID / 128), 256, 0, stream>>>(
        yb, w1t, b1, nullptr, hb, MROWS, HID, EE, 1);

    // MLP down: h @ W2 + b2 -> f32 [4096,768]
    k_gemm<0><<<dim3(MROWS / 64, EE / 128), 256, 0, stream>>>(
        hb, w2t, b2, mlpf, nullptr, MROWS, EE, HID, 0);

    k_ln<<<MROWS, 256, 0, stream>>>(yf, mlpf, ln2_g, ln2_b, out, nullptr);
}
